// EnhancedHelformer_45019847196809
// MI455X (gfx1250) — matrix operand order repaired and checked by reading, not compiled, not run on hardware
//
#include <hip/hip_runtime.h>
#include <hip/hip_bf16.h>

// ---------------------------------------------------------------------------
// EnhancedHelformer forward for MI455X (gfx1250, wave32, WMMA + TDM).
// Hot GEMMs: double-buffered tensor_load_to_lds (TDM) overlapped with
// v_wmma_f32_16x16x32_bf16 (bf16 in, f32 acc). Edge/odd shapes: manual staging.
// ---------------------------------------------------------------------------

typedef __attribute__((ext_vector_type(16))) __bf16 bf16x16;
typedef __attribute__((ext_vector_type(8)))  __bf16 bf16x8;
typedef __attribute__((ext_vector_type(8)))  float  f32x8;

union BF16Frag {
    bf16x16 v;
    struct Halves { bf16x8 lo; bf16x8 hi; } h;
};

#define D_MODEL 384
#define SEQ 512
#define NB 4
#define NH 16
#define HD 24

#ifndef __has_builtin
#define __has_builtin(x) 0
#endif
#if __has_builtin(__builtin_amdgcn_tensor_load_to_lds)
#define HAS_TDM 1
#else
#define HAS_TDM 0
#endif

#if HAS_TDM
typedef unsigned int u32x4 __attribute__((ext_vector_type(4)));
typedef int          i32x8 __attribute__((ext_vector_type(8)));
typedef int          i32x4 __attribute__((ext_vector_type(4)));

// TDM 2D load of a (rows x 32) bf16 tile into LDS with row padding that matches
// our [rows][40] __bf16 LDS arrays (64B data + 16B pad per row).
// tensor dims == tile dims (caller guarantees fully in-bounds tile).
__device__ __forceinline__ void tdm_load_tile_b16(
    unsigned lds_byte, const void* gptr, unsigned rows, unsigned stride_elems)
{
    unsigned long long ga = (unsigned long long)gptr;
    u32x4 g0;
    g0.x = 1u;                                                  // count=1 (valid user D#)
    g0.y = lds_byte;                                            // lds_addr (bytes)
    g0.z = (unsigned)(ga & 0xFFFFFFFFull);                      // global_addr[31:0]
    g0.w = (unsigned)((ga >> 32) & 0x1FFFFFFull) | (2u << 30);  // [56:32] | type=2
    i32x8 g1;
    // data_size=2B | pad_enable | pad_interval=16DW (64B) | pad_amount=4DW (16B)
    g1[0] = (int)((1u << 16) | (1u << 20) | (3u << 22) | (3u << 25));
    g1[1] = (int)(32u << 16);                // tensor_dim0[15:0]=32 at bits 63:48
    g1[2] = (int)((rows & 0xFFFFu) << 16);   // tensor_dim0 hi=0 | tensor_dim1 lo
    g1[3] = (int)(32u << 16);                // tensor_dim1 hi=0 | tile_dim0=32
    g1[4] = (int)(rows & 0xFFFFu);           // tile_dim1 | tile_dim2=0
    g1[5] = (int)stride_elems;               // tensor_dim0_stride[31:0] (elems)
    g1[6] = 0;                               // stride hi | dim1_stride lo
    g1[7] = 0;
    i32x4 z4 = {0, 0, 0, 0};
#if __clang_major__ >= 23
    i32x8 z8 = {0, 0, 0, 0, 0, 0, 0, 0};
    __builtin_amdgcn_tensor_load_to_lds(g0, g1, z4, z4, z8, 0);
#else
    __builtin_amdgcn_tensor_load_to_lds(g0, g1, z4, z4, 0);
#endif
}

__device__ __forceinline__ void tdm_wait()
{
#if __has_builtin(__builtin_amdgcn_s_wait_tensorcnt)
    __builtin_amdgcn_s_wait_tensorcnt(0);
#else
    asm volatile("s_wait_tensorcnt 0x0" ::: "memory");
#endif
}

__device__ __forceinline__ unsigned lds_off(const void* p)
{
    // Generic LDS addresses carry the LDS offset in the low 32 bits (ISA 10.2).
    return (unsigned)(unsigned long long)p;
}
#endif // HAS_TDM

// One K=32 step: load fragments per ISA lane layout, 4 WMMAs into acc[2][2].
__device__ __forceinline__ void wmma_step(
    const __bf16 (*__restrict__ As)[40], const __bf16 (*__restrict__ Bs)[40],
    int wm, int wn, int half, int lr, f32x8 acc[2][2])
{
    const int r0 = wm * 32, c0 = wn * 32;
    BF16Frag a0, a1, b0, b1;
    const int akb = half * 8;   // A: lanes16-31 hold K 8..15 / 24..31
    a0.h.lo = *(const bf16x8*)&As[r0 + lr][akb];
    a0.h.hi = *(const bf16x8*)&As[r0 + lr][akb + 16];
    a1.h.lo = *(const bf16x8*)&As[r0 + 16 + lr][akb];
    a1.h.hi = *(const bf16x8*)&As[r0 + 16 + lr][akb + 16];
    const int bkb = half * 8;   // B: lanes16-31 hold K 8..15 / 24..31 of same col
    b0.h.lo = *(const bf16x8*)&Bs[c0 + lr][bkb];
    b0.h.hi = *(const bf16x8*)&Bs[c0 + lr][bkb + 16];
    b1.h.lo = *(const bf16x8*)&Bs[c0 + 16 + lr][bkb];
    b1.h.hi = *(const bf16x8*)&Bs[c0 + 16 + lr][bkb + 16];
    acc[0][0] = __builtin_amdgcn_wmma_f32_16x16x32_bf16(false, a0.v, false, b0.v, (short)0, acc[0][0], false, false);
    acc[0][1] = __builtin_amdgcn_wmma_f32_16x16x32_bf16(false, a0.v, false, b1.v, (short)0, acc[0][1], false, false);
    acc[1][0] = __builtin_amdgcn_wmma_f32_16x16x32_bf16(false, a1.v, false, b0.v, (short)0, acc[1][0], false, false);
    acc[1][1] = __builtin_amdgcn_wmma_f32_16x16x32_bf16(false, a1.v, false, b1.v, (short)0, acc[1][1], false, false);
}

// ---------------------------------------------------------------------------
// Generic batched GEMM: C[z] = act(A[z] @ op(B[z]) + bias) * scale  (+= opt)
// A: [M,K] bf16 row-major (lda). BNT: B is [N,K] (weight layout, computes A@B^T).
// !BNT: B is [K,N]. C f32 [M,N]; optional bf16 mirror.
// Block: 256 thr = 8 waves, tile 128(M) x 64(N), wave tile 32x32 (2x2 WMMA).
// Fast path: double-buffered TDM staging overlapped with WMMA.
// ---------------------------------------------------------------------------
template<bool BNT>
__global__ __launch_bounds__(256) void k_gemm(
    const __bf16* __restrict__ Abase, long lda, long zA,
    const __bf16* __restrict__ Bbase, long ldb, long zB,
    float* __restrict__ Cbase, long ldc, long zC,
    __bf16* __restrict__ Cbfbase, long ldcb, long zCb,
    const float* __restrict__ bias,
    int M, int N, int K, int act, float scale, int accumulate)
{
    __shared__ alignas(16) __bf16 As[2][128][40];
    __shared__ alignas(16) __bf16 Bs[2][64][40];
    const int tid = threadIdx.x;
    const int z = blockIdx.z;
    const __bf16* A = Abase + (long)z * zA;
    const __bf16* B = Bbase + (long)z * zB;
    float* C = Cbase + (long)z * zC;
    __bf16* Cb = Cbfbase ? (Cbfbase + (long)z * zCb) : (__bf16*)0;
    const int bn0 = blockIdx.x * 64;
    const int bm0 = blockIdx.y * 128;
    const int wave = tid >> 5, lane = tid & 31;
    const int wm = wave & 3, wn = wave >> 2;
    const int half = lane >> 4, lr = lane & 15;

    f32x8 acc[2][2] = {};

#if HAS_TDM
    const bool fast = BNT && (bm0 + 128 <= M) && (bn0 + 64 <= N) && ((K & 31) == 0);
    if (fast) {
        // ---- prologue: DMA first tiles
        if (wave == 0) {
            tdm_load_tile_b16(lds_off(&As[0][0][0]), A + (long)bm0 * lda, 128, (unsigned)lda);
            tdm_load_tile_b16(lds_off(&Bs[0][0][0]), B + (long)bn0 * ldb, 64, (unsigned)ldb);
            tdm_wait();
        }
        __syncthreads();
        int cur = 0;
        for (int k0 = 0; k0 < K; k0 += 32) {
            const int nxt = cur ^ 1;
            if (wave == 0 && (k0 + 32) < K) {   // DMA next tiles while computing
                tdm_load_tile_b16(lds_off(&As[nxt][0][0]),
                                  A + (long)bm0 * lda + k0 + 32, 128, (unsigned)lda);
                tdm_load_tile_b16(lds_off(&Bs[nxt][0][0]),
                                  B + (long)bn0 * ldb + k0 + 32, 64, (unsigned)ldb);
            }
            wmma_step(As[cur], Bs[cur], wm, wn, half, lr, acc);
            if (wave == 0) tdm_wait();
            __syncthreads();
            cur = nxt;
        }
    } else
#endif
    {
        for (int k0 = 0; k0 < K; k0 += 32) {
            // ---- stage A tile (128x32) : one 16-elem half-row per thread
            {
                int row = tid >> 1, kh = (tid & 1) * 16;
                int m = bm0 + row;
                __bf16* dst = &As[0][row][kh];
                if (m < M && (k0 + kh + 16) <= K) {
                    const uint4* src = (const uint4*)(A + (long)m * lda + k0 + kh);
                    *(uint4*)dst = src[0];
                    *(uint4*)(dst + 8) = src[1];
                } else {
                    for (int j = 0; j < 16; ++j) {
                        int kk = k0 + kh + j;
                        dst[j] = (m < M && kk < K) ? A[(long)m * lda + kk] : (__bf16)0.0f;
                    }
                }
            }
            // ---- stage B tile (64x32 as [n][k])
            if (BNT) {
                int row = tid >> 2, kq = (tid & 3) * 8;
                int n = bn0 + row;
                __bf16* dst = &Bs[0][row][kq];
                if (n < N && (k0 + kq + 8) <= K) {
                    *(uint4*)dst = *(const uint4*)(B + (long)n * ldb + k0 + kq);
                } else {
                    for (int j = 0; j < 8; ++j) {
                        int kk = k0 + kq + j;
                        dst[j] = (n < N && kk < K) ? B[(long)n * ldb + kk] : (__bf16)0.0f;
                    }
                }
            } else {
                int n = tid & 63, kq = (tid >> 6) * 8;
                for (int j = 0; j < 8; ++j) {
                    int kk = k0 + kq + j;
                    Bs[0][n][kq + j] = ((bn0 + n) < N && kk < K)
                                     ? B[(long)kk * ldb + bn0 + n] : (__bf16)0.0f;
                }
            }
            __syncthreads();
            wmma_step(As[0], Bs[0], wm, wn, half, lr, acc);
            __syncthreads();
        }
    }

    // ---- epilogue (C/D layout: lane<16 -> rows 0..7, lane>=16 -> rows 8..15)
    for (int i = 0; i < 2; ++i)
        for (int j = 0; j < 2; ++j)
            for (int r = 0; r < 8; ++r) {
                int m = bm0 + wm * 32 + i * 16 + half * 8 + r;
                int n = bn0 + wn * 32 + j * 16 + lr;
                if (m < M && n < N) {
                    float v = acc[i][j][r];
                    if (bias) v += bias[n];
                    if (act == 1) v = 0.5f * v * (1.0f + erff(v * 0.70710678118f)); // exact gelu
                    else if (act == 2) v = fmaxf(v, 0.0f);
                    v *= scale;
                    float outv = accumulate ? (C[(long)m * ldc + n] + v) : v;
                    C[(long)m * ldc + n] = outv;
                    if (Cb) Cb[(long)m * ldcb + n] = (__bf16)outv;
                }
            }
}

// ---------------------------------------------------------------------------
// Depthconv as shifted GEMM: Y[b,t,:] = sum_tap X[b,t-pad+tap,:] @ Wt[tap]^T + b
// X: [4*512, 384] bf16. Wt: [ksz][384out][384in] bf16. Y: [4*L, 384] f32.
// Weight tiles via double-buffered TDM; activation tiles manually staged
// (temporal shift + zero padding) into the ping-pong buffer.
// ---------------------------------------------------------------------------
__global__ __launch_bounds__(256) void k_conv(
    const __bf16* __restrict__ X, const __bf16* __restrict__ Wt,
    const float* __restrict__ bias, float* __restrict__ Y,
    int ksz, int pad, int L)
{
    __shared__ alignas(16) __bf16 As[2][128][40];
    __shared__ alignas(16) __bf16 Bs[2][64][40];
    const int tid = threadIdx.x;
    const int bn0 = blockIdx.x * 64;
    const int bm0 = blockIdx.y * 128;
    const int Mrows = NB * L;
    const int wave = tid >> 5, lane = tid & 31;
    const int wm = wave & 3, wn = wave >> 2;
    const int half = lane >> 4, lr = lane & 15;
    f32x8 acc[2][2] = {};

    // stage A tile for (tap, k0) into buffer `buf`
    auto stageA = [&](int buf, int tap, int k0) {
        int row = tid >> 1, kh = (tid & 1) * 16;
        int r = bm0 + row;
        __bf16* dst = &As[buf][row][kh];
        int b = r / L, t = r - b * L;
        int tin = t - pad + tap;
        if (r < Mrows && tin >= 0 && tin < SEQ) {
            const uint4* src = (const uint4*)(X + ((long)(b * SEQ + tin)) * D_MODEL + k0 + kh);
            *(uint4*)dst = src[0];
            *(uint4*)(dst + 8) = src[1];
        } else {
            for (int j = 0; j < 16; ++j) dst[j] = (__bf16)0.0f;
        }
    };

    const int KSTEPS = D_MODEL / 32;            // 12
    const int total = ksz * KSTEPS;

#if HAS_TDM
    stageA(0, 0, 0);
    if (wave == 0) {
        tdm_load_tile_b16(lds_off(&Bs[0][0][0]), Wt + (long)bn0 * D_MODEL, 64, D_MODEL);
        tdm_wait();
    }
    __syncthreads();
    int cur = 0;
    for (int it = 0; it < total; ++it) {
        const int nxt = cur ^ 1;
        const int nit = it + 1;
        if (nit < total) {
            const int ntap = nit / KSTEPS, nk0 = (nit % KSTEPS) * 32;
            stageA(nxt, ntap, nk0);
            if (wave == 0)
                tdm_load_tile_b16(lds_off(&Bs[nxt][0][0]),
                                  Wt + (long)ntap * D_MODEL * D_MODEL
                                     + (long)bn0 * D_MODEL + nk0, 64, D_MODEL);
        }
        wmma_step(As[cur], Bs[cur], wm, wn, half, lr, acc);
        if (wave == 0) tdm_wait();
        __syncthreads();
        cur = nxt;
    }
#else
    for (int it = 0; it < total; ++it) {
        const int tap = it / KSTEPS, k0 = (it % KSTEPS) * 32;
        stageA(0, tap, k0);
        { // stage B (weights, [N,K] layout; grid covers N=384 exactly)
            int row = tid >> 2, kq = (tid & 3) * 8;
            int n = bn0 + row;
            *(uint4*)&Bs[0][row][kq] = *(const uint4*)(
                Wt + (long)tap * D_MODEL * D_MODEL + (long)n * D_MODEL + k0 + kq);
        }
        __syncthreads();
        wmma_step(As[0], Bs[0], wm, wn, half, lr, acc);
        __syncthreads();
    }
#endif
    for (int i = 0; i < 2; ++i)
        for (int j = 0; j < 2; ++j)
            for (int r = 0; r < 8; ++r) {
                int m = bm0 + wm * 32 + i * 16 + half * 8 + r;
                int n = bn0 + wn * 32 + j * 16 + lr;
                if (m < Mrows) Y[(long)m * D_MODEL + n] = acc[i][j][r] + bias[n];
            }
}

// ---------------------------------------------------------------------------
// Row softmax: probs(bf16) = softmax(scores*scale + causal_mask). row = pair*S+q
// ---------------------------------------------------------------------------
__global__ __launch_bounds__(256) void k_softmax(
    const float* __restrict__ Sc, __bf16* __restrict__ P,
    int Scols, float scale, int causal)
{
    const int row = blockIdx.x;
    const int q = row % Scols;
    const float* src = Sc + (long)row * Scols;
    __bf16* dst = P + (long)row * Scols;
    __shared__ float red[256];
    const int valid = causal ? (q + 1) : Scols;
    float mx = -1e30f;
    for (int c = threadIdx.x; c < valid; c += 256) mx = fmaxf(mx, src[c] * scale);
    red[threadIdx.x] = mx; __syncthreads();
    for (int s = 128; s > 0; s >>= 1) {
        if (threadIdx.x < s) red[threadIdx.x] = fmaxf(red[threadIdx.x], red[threadIdx.x + s]);
        __syncthreads();
    }
    mx = red[0]; __syncthreads();
    float sum = 0.0f;
    for (int c = threadIdx.x; c < valid; c += 256) sum += expf(src[c] * scale - mx);
    red[threadIdx.x] = sum; __syncthreads();
    for (int s = 128; s > 0; s >>= 1) {
        if (threadIdx.x < s) red[threadIdx.x] += red[threadIdx.x + s];
        __syncthreads();
    }
    const float inv = 1.0f / red[0];
    for (int c = threadIdx.x; c < Scols; c += 256) {
        float p = (c < valid) ? expf(src[c] * scale - mx) * inv : 0.0f;
        dst[c] = (__bf16)p;
    }
}

// ---------------------------------------------------------------------------
// x = LN(x (+ residual)) * g + b ; also refresh bf16 mirror. D=384, 128 thr.
// ---------------------------------------------------------------------------
__global__ __launch_bounds__(128) void k_add_ln(
    float* __restrict__ X, const float* __restrict__ R,
    const float* __restrict__ g, const float* __restrict__ bb,
    __bf16* __restrict__ Xbf)
{
    const long row = blockIdx.x;
    float* x = X + row * D_MODEL;
    const float* r = R ? (R + row * D_MODEL) : (const float*)0;
    __shared__ float red[128];
    float v[3];
    float s = 0.0f;
    for (int i = 0; i < 3; ++i) {
        int d = threadIdx.x + i * 128;
        float t = x[d] + (r ? r[d] : 0.0f);
        v[i] = t; s += t;
    }
    red[threadIdx.x] = s; __syncthreads();
    for (int st = 64; st > 0; st >>= 1) {
        if (threadIdx.x < st) red[threadIdx.x] += red[threadIdx.x + st];
        __syncthreads();
    }
    const float mean = red[0] / (float)D_MODEL; __syncthreads();
    float vs = 0.0f;
    for (int i = 0; i < 3; ++i) { float d = v[i] - mean; vs += d * d; }
    red[threadIdx.x] = vs; __syncthreads();
    for (int st = 64; st > 0; st >>= 1) {
        if (threadIdx.x < st) red[threadIdx.x] += red[threadIdx.x + st];
        __syncthreads();
    }
    const float rstd = rsqrtf(red[0] / (float)D_MODEL + 1e-5f);
    for (int i = 0; i < 3; ++i) {
        int d = threadIdx.x + i * 128;
        float o = (v[i] - mean) * rstd * g[d] + bb[d];
        x[d] = o;
        if (Xbf) Xbf[row * D_MODEL + d] = (__bf16)o;
    }
}

// --------------------------- small fp32 kernels ----------------------------
__global__ void k_cvt(const float* __restrict__ src, __bf16* __restrict__ dst, long n)
{
    long i = (long)blockIdx.x * blockDim.x + threadIdx.x;
    if (i < n) dst[i] = (__bf16)src[i];
}

// w: [384out][384in][ksz] -> wt: [ksz][384out][384in] bf16
__global__ void k_cvt_convw(const float* __restrict__ w, __bf16* __restrict__ wt, int ksz)
{
    long n = (long)D_MODEL * D_MODEL * ksz;
    long i = (long)blockIdx.x * blockDim.x + threadIdx.x;
    if (i >= n) return;
    long tap = i / (D_MODEL * D_MODEL);
    long rem = i - tap * (D_MODEL * D_MODEL);
    int o = (int)(rem / D_MODEL), in = (int)(rem % D_MODEL);
    wt[i] = (__bf16)w[((long)o * D_MODEL + in) * ksz + tap];
}

__global__ void k_embed(
    const float* __restrict__ price, const float* __restrict__ volume,
    const float* __restrict__ vola, const float* __restrict__ sent,
    const int* __restrict__ hour, const int* __restrict__ day,
    const float* pw, const float* pb, const float* vw, const float* vb,
    const float* ow, const float* ob, const float* sw, const float* sb,
    const float* temb, const float* demb,
    __bf16* __restrict__ comb) // [2048, 576]
{
    const int row = blockIdx.x;
    for (int c = threadIdx.x; c < 576; c += blockDim.x) {
        int piece = c / 96, j = c % 96;
        float val;
        switch (piece) {
            case 0:  val = price[row]  * pw[j] + pb[j]; break;
            case 1:  val = volume[row] * vw[j] + vb[j]; break;
            case 2:  val = vola[row]   * ow[j] + ob[j]; break;
            case 3:  val = sent[row]   * sw[j] + sb[j]; break;
            case 4:  val = temb[hour[row] * 96 + j];    break;
            default: val = demb[day[row]  * 96 + j];    break;
        }
        comb[(long)row * 576 + c] = (__bf16)val;
    }
}

__global__ void k_hw_scan(const float* __restrict__ price,
                          const float* alpha, const float* beta, const float* gamma,
                          float* __restrict__ lv, float* __restrict__ tr,
                          float* __restrict__ se)
{
    const int b = threadIdx.x;
    __shared__ float shist[NB][24];
    if (b >= NB) return;
    const float a  = 1.0f / (1.0f + expf(-alpha[0]));
    const float bt = 1.0f / (1.0f + expf(-beta[0]));
    const float g  = 1.0f / (1.0f + expf(-gamma[0]));
    for (int i = 0; i < 24; ++i) shist[b][i] = 0.0f;
    float level = price[b * SEQ];
    float trend = 0.0f;
    lv[b * SEQ] = level; tr[b * SEQ] = 0.0f; se[b * SEQ] = 0.0f;
    for (int t = 1; t < SEQ; ++t) {
        int idx = t % 24;
        float prev = (t >= 24) ? shist[b][idx] : 0.0f;
        float x = price[b * SEQ + t];
        float nl = a * (x - prev) + (1.0f - a) * (level + trend);
        float nt = bt * (nl - level) + (1.0f - bt) * trend;
        float ns = g * (x - nl) + (1.0f - g) * prev;
        shist[b][idx] = ns;
        level = nl; trend = nt;
        lv[b * SEQ + t] = nl; tr[b * SEQ + t] = nt; se[b * SEQ + t] = ns;
    }
}

__global__ void k_hw_embed(const float* __restrict__ lv, const float* __restrict__ tr,
                           const float* __restrict__ se,
                           const float* lw, const float* lb, const float* tw, const float* tb,
                           const float* sw, const float* sb,
                           __bf16* __restrict__ out) // [2048, 384]
{
    const int row = blockIdx.x;
    for (int c = threadIdx.x; c < D_MODEL; c += blockDim.x) {
        int piece = c / 128, j = c % 128;
        float src = (piece == 0) ? lv[row] : (piece == 1) ? tr[row] : se[row];
        const float* w = (piece == 0) ? lw : (piece == 1) ? tw : sw;
        const float* bb = (piece == 0) ? lb : (piece == 1) ? tb : sb;
        out[(long)row * D_MODEL + c] = (__bf16)(src * w[j] + bb[j]);
    }
}

__global__ void k_posenc(float* __restrict__ h, __bf16* __restrict__ hbf)
{
    const int row = blockIdx.x;
    const int s = row % SEQ;
    for (int d = threadIdx.x; d < D_MODEL; d += blockDim.x) {
        int i = d >> 1;
        float div = expf((float)(2 * i) * (-9.210340371976184f / (float)D_MODEL));
        float ang = (float)s * div;
        float pe = (d & 1) ? cosf(ang) : sinf(ang);
        float v = h[(long)row * D_MODEL + d] + pe;
        h[(long)row * D_MODEL + d] = v;
        hbf[(long)row * D_MODEL + d] = (__bf16)v;
    }
}

// y: [4*L, 384] f32 -> ms_bf row ((b*4+scaleIdx)*512+s) interpolated to S=512
__global__ void k_interp(const float* __restrict__ y, __bf16* __restrict__ ms,
                         int L, int scaleIdx)
{
    const int row = blockIdx.x;           // b*512+s
    const int b = row >> 9, s = row & 511;
    float src = (s + 0.5f) * ((float)L / (float)SEQ) - 0.5f;
    src = fminf(fmaxf(src, 0.0f), (float)(L - 1));
    int i0 = (int)src;
    int i1 = i0 + 1; if (i1 > L - 1) i1 = L - 1;
    float w = src - (float)i0;
    const float* y0 = y + ((long)b * L + i0) * D_MODEL;
    const float* y1 = y + ((long)b * L + i1) * D_MODEL;
    __bf16* dst = ms + (((long)b * 4 + scaleIdx) * SEQ + s) * D_MODEL;
    for (int d = threadIdx.x; d < D_MODEL; d += blockDim.x)
        dst[d] = (__bf16)(y0[d] * (1.0f - w) + y1[d] * w);
}

__global__ __launch_bounds__(128) void k_pool(const float* __restrict__ h,
                                              float* __restrict__ pooled) // [4,768]
{
    const int b = blockIdx.x / D_MODEL, d = blockIdx.x % D_MODEL;
    __shared__ float rm[128], rx[128];
    float sm = 0.0f, mx = -1e30f;
    for (int s = threadIdx.x; s < SEQ; s += 128) {
        float v = h[((long)b * SEQ + s) * D_MODEL + d];
        sm += v; mx = fmaxf(mx, v);
    }
    rm[threadIdx.x] = sm; rx[threadIdx.x] = mx; __syncthreads();
    for (int st = 64; st > 0; st >>= 1) {
        if (threadIdx.x < st) {
            rm[threadIdx.x] += rm[threadIdx.x + st];
            rx[threadIdx.x] = fmaxf(rx[threadIdx.x], rx[threadIdx.x + st]);
        }
        __syncthreads();
    }
    if (threadIdx.x == 0) {
        pooled[b * 768 + d] = rm[0] / (float)SEQ;
        pooled[b * 768 + D_MODEL + d] = rx[0];
    }
}

__global__ __launch_bounds__(256) void k_heads(
    const float* __restrict__ pooled,
    const float* p1w, const float* p1b, const float* p2w, const float* p2b,
    const float* p3w, const float* p3b,
    const float* q1w, const float* q1b, const float* q2w, const float* q2b,
    const float* c1w, const float* c1b, const float* c2w, const float* c2b,
    float* __restrict__ out)
{
    __shared__ float z1[NB][384];
    __shared__ float z2[NB][192];
    __shared__ float qh[NB][3][192];
    __shared__ float ch[NB][192];
    const int tid = threadIdx.x;
    for (int i = tid; i < NB * 384; i += 256) {
        int b = i / 384, o = i % 384;
        float s = p1b[o];
        const float* pr = pooled + b * 768;
        const float* w = p1w + (long)o * 768;
        for (int k = 0; k < 768; ++k) s += pr[k] * w[k];
        z1[b][o] = fmaxf(s, 0.0f);
    }
    __syncthreads();
    for (int i = tid; i < NB * 192; i += 256) {
        int b = i / 192, o = i % 192;
        float s = p2b[o];
        const float* w = p2w + (long)o * 384;
        for (int k = 0; k < 384; ++k) s += z1[b][k] * w[k];
        z2[b][o] = fmaxf(s, 0.0f);
    }
    for (int i = tid; i < NB * 3 * 192; i += 256) {
        int b = i / (3 * 192), r = i % (3 * 192);
        int q = r / 192, k = r % 192;
        float s = q1b[q * 192 + k];
        const float* w = q1w + ((long)q * 192 + k) * 384;
        const float* gp = pooled + b * 768;
        for (int d = 0; d < 384; ++d) s += gp[d] * w[d];
        qh[b][q][k] = fmaxf(s, 0.0f);
    }
    for (int i = tid; i < NB * 192; i += 256) {
        int b = i / 192, k = i % 192;
        float s = c1b[k];
        const float* w = c1w + (long)k * 384;
        const float* gp = pooled + b * 768;
        for (int d = 0; d < 384; ++d) s += gp[d] * w[d];
        ch[b][k] = fmaxf(s, 0.0f);
    }
    __syncthreads();
    if (tid < NB) {
        const int b = tid;
        for (int o = 0; o < 3; ++o) {
            float s = p3b[o];
            for (int k = 0; k < 192; ++k) s += z2[b][k] * p3w[o * 192 + k];
            out[b * 7 + o] = s;
        }
        float qv[3];
        for (int q = 0; q < 3; ++q) {
            float s = q2b[q];
            for (int k = 0; k < 192; ++k) s += qh[b][q][k] * q2w[q * 192 + k];
            qv[q] = s;
        }
        float t;
        if (qv[0] > qv[1]) { t = qv[0]; qv[0] = qv[1]; qv[1] = t; }
        if (qv[1] > qv[2]) { t = qv[1]; qv[1] = qv[2]; qv[2] = t; }
        if (qv[0] > qv[1]) { t = qv[0]; qv[0] = qv[1]; qv[1] = t; }
        out[b * 7 + 3] = qv[0]; out[b * 7 + 4] = qv[1]; out[b * 7 + 5] = qv[2];
        float s = c2b[0];
        for (int k = 0; k < 192; ++k) s += ch[b][k] * c2w[k];
        out[b * 7 + 6] = 1.0f / (1.0f + expf(-s));
    }
}

// ---------------------------------------------------------------------------
// Host orchestration
// ---------------------------------------------------------------------------
extern "C" void kernel_launch(void* const* d_in, const int* in_sizes, int n_in,
                              void* d_out, int out_size, void* d_ws, size_t ws_size,
                              hipStream_t stream)
{
    (void)in_sizes; (void)n_in; (void)out_size; (void)ws_size;
    auto F = [&](int i) { return (const float*)d_in[i]; };
    auto Ii = [&](int i) { return (const int*)d_in[i]; };

    // ---- workspace bump allocator
    size_t off = 0;
    auto alloc = [&](size_t bytes) -> void* {
        off = (off + 255) & ~(size_t)255;
        void* p = (char*)d_ws + off;
        off += bytes;
        return p;
    };
    const long RWS = 2048;                       // B*S rows
    float*  h       = (float*) alloc(RWS * 384 * 4);
    __bf16* h_bf    = (__bf16*)alloc(RWS * 384 * 2);
    __bf16* comb_bf = (__bf16*)alloc(RWS * 576 * 2);
    float*  tmpA    = (float*) alloc(RWS * 1536 * 4);
    __bf16* tmpA_bf = (__bf16*)alloc(RWS * 1536 * 2);
    float*  tmpB    = (float*) alloc(RWS * 384 * 4);
    float*  attnb   = (float*) alloc(RWS * 384 * 4);
    __bf16* attnb_bf= (__bf16*)alloc(RWS * 384 * 2);
    float*  scores  = (float*) alloc(16L * SEQ * SEQ * 4);
    __bf16* probs   = (__bf16*)alloc(16L * SEQ * SEQ * 2);
    float*  ybuf    = (float*) alloc(4L * 513 * 384 * 4);
    __bf16* ms_bf   = (__bf16*)alloc(16L * SEQ * 384 * 2);
    float*  qf      = (float*) alloc(RWS * 384 * 4);
    __bf16* qf_bf   = (__bf16*)alloc(RWS * 384 * 2);
    float*  kvf     = (float*) alloc(16L * SEQ * 768 * 4);
    __bf16* kvf_bf  = (__bf16*)alloc(16L * SEQ * 768 * 2);
    float*  fub     = (float*) alloc((long)SEQ * 384 * 4);
    __bf16* fub_bf  = (__bf16*)alloc((long)SEQ * 384 * 2);
    float*  msum    = (float*) alloc(RWS * 384 * 4);
    float*  lv      = (float*) alloc(RWS * 4);
    float*  tr      = (float*) alloc(RWS * 4);
    float*  se      = (float*) alloc(RWS * 4);
    __bf16* hwv_bf  = (__bf16*)alloc(RWS * 384 * 2);
    float*  pooled  = (float*) alloc(4 * 768 * 4);
    __bf16* wbuf    = (__bf16*)alloc(168L * 384 * 384 * 2);  // weight staging

    auto cvt = [&](const float* src, __bf16* dst, long n) {
        k_cvt<<<dim3((unsigned)((n + 255) / 256)), 256, 0, stream>>>(src, dst, n);
    };
    auto gemm = [&](const __bf16* A, long lda, long zA,
                    const __bf16* B, long ldb, long zB,
                    float* C, long ldc, long zC,
                    __bf16* Cb, long ldcb, long zCb,
                    const float* bias, int M, int N, int K, int Z,
                    int act, float scale, int accum, bool bnt) {
        dim3 grid((N + 63) / 64, (M + 127) / 128, Z);
        if (bnt)
            k_gemm<true><<<grid, 256, 0, stream>>>(A, lda, zA, B, ldb, zB, C, ldc, zC,
                                                   Cb, ldcb, zCb, bias, M, N, K, act, scale, accum);
        else
            k_gemm<false><<<grid, 256, 0, stream>>>(A, lda, zA, B, ldb, zB, C, ldc, zC,
                                                    Cb, ldcb, zCb, bias, M, N, K, act, scale, accum);
    };

    const float ATT_SCALE = 0.20412414523193154f; // 1/sqrt(24)

    // ---------------- embeddings ----------------
    k_embed<<<RWS, 192, 0, stream>>>((const float*)d_in[0], F(1), F(2), F(3),
                                     Ii(4), Ii(5),
                                     F(6), F(7), F(8), F(9), F(10), F(11), F(12), F(13),
                                     F(14), F(15), comb_bf);
    cvt(F(16), wbuf, 384L * 576);
    gemm(comb_bf, 576, 0, wbuf, 576, 0, h, 384, 0, (__bf16*)0, 0, 0,
         F(17), 2048, 384, 576, 1, 0, 1.0f, 0, true);
    k_add_ln<<<RWS, 128, 0, stream>>>(h, (const float*)0, F(18), F(19), h_bf);

    // ---------------- Holt-Winters ----------------
    k_hw_scan<<<1, 32, 0, stream>>>((const float*)d_in[0], F(20), F(21), F(22), lv, tr, se);
    k_hw_embed<<<RWS, 128, 0, stream>>>(lv, tr, se, F(23), F(24), F(25), F(26), F(27), F(28), hwv_bf);
    cvt(F(29), wbuf, 384L * 384);
    gemm(hwv_bf, 384, 0, wbuf, 384, 0, h, 384, 0, (__bf16*)0, 0, 0,
         F(30), 2048, 384, 384, 1, 0, 1.0f, 1, true);
    k_posenc<<<RWS, 128, 0, stream>>>(h, h_bf);

    // ---------------- transformer blocks ----------------
    const int kk[4] = {1, 4, 24, 168};
    const int wi[4] = {4, 6, 8, 10};
    for (int l = 0; l < 4; ++l) {
        const int bi = 31 + l * 26;

        // self-attention QKV
        cvt(F(bi + 0), wbuf, 1152L * 384);
        gemm(h_bf, 384, 0, wbuf, 384, 0, tmpA, 1152, 0, tmpA_bf, 1152, 0,
             F(bi + 1), 2048, 1152, 384, 1, 0, 1.0f, 0, true);
        for (int b = 0; b < NB; ++b) {
            const __bf16* base = tmpA_bf + (long)b * SEQ * 1152;
            gemm(base, 1152, 24, base + 384, 1152, 24,
                 scores, SEQ, (long)SEQ * SEQ, (__bf16*)0, 0, 0,
                 (const float*)0, SEQ, SEQ, HD, NH, 0, 1.0f, 0, true);
            k_softmax<<<NH * SEQ, 256, 0, stream>>>(scores, probs, SEQ, ATT_SCALE, 1);
            gemm(probs, SEQ, (long)SEQ * SEQ, base + 768, 1152, 24,
                 attnb + (long)b * SEQ * 384, 384, 24,
                 attnb_bf + (long)b * SEQ * 384, 384, 24,
                 (const float*)0, SEQ, HD, SEQ, NH, 0, 1.0f, 0, false);
        }
        cvt(F(bi + 2), wbuf, 384L * 384);
        gemm(attnb_bf, 384, 0, wbuf, 384, 0, tmpB, 384, 0, (__bf16*)0, 0, 0,
             F(bi + 3), 2048, 384, 384, 1, 0, 1.0f, 0, true);
        k_add_ln<<<RWS, 128, 0, stream>>>(h, tmpB, F(bi + 20), F(bi + 21), h_bf);

        // multi-scale convs -> ms_bf
        for (int sidx = 0; sidx < 4; ++sidx) {
            int ks = kk[sidx];
            int L = (ks == 1) ? 512 : 513;
            long nw = 384L * 384 * ks;
            k_cvt_convw<<<dim3((unsigned)((nw + 255) / 256)), 256, 0, stream>>>(F(bi + wi[sidx]), wbuf, ks);
            dim3 cg(384 / 64, (unsigned)((4 * L + 127) / 128));
            k_conv<<<cg, 256, 0, stream>>>(h_bf, wbuf, F(bi + wi[sidx] + 1), ybuf, ks, ks / 2, L);
            k_interp<<<RWS, 128, 0, stream>>>(ybuf, ms_bf, L, sidx);
        }

        // fusion attention (q from h tiled, kv from ms)
        cvt(F(bi + 12), wbuf, 1152L * 384);
        gemm(h_bf, 384, 0, wbuf, 384, 0, qf, 384, 0, qf_bf, 384, 0,
             F(bi + 13), 2048, 384, 384, 1, 0, 1.0f, 0, true);
        gemm(ms_bf, 384, 0, wbuf + 384L * 384, 384, 0, kvf, 768, 0, kvf_bf, 768, 0,
             F(bi + 13) + 384, 8192, 768, 384, 1, 0, 1.0f, 0, true);
        cvt(F(bi + 14), wbuf, 384L * 384);     // fus_out (fus_in no longer needed)
        for (int bn = 0; bn < 16; ++bn) {
            const __bf16* qn = qf_bf + (long)(bn & 3) * SEQ * 384;   // q = tile(x) -> x[bn%4]
            gemm(qn, 384, 24, kvf_bf + (long)bn * SEQ * 768, 768, 24,
                 scores, SEQ, (long)SEQ * SEQ, (__bf16*)0, 0, 0,
                 (const float*)0, SEQ, SEQ, HD, NH, 0, 1.0f, 0, true);
            k_softmax<<<NH * SEQ, 256, 0, stream>>>(scores, probs, SEQ, ATT_SCALE, 0);
            gemm(probs, SEQ, (long)SEQ * SEQ, kvf_bf + (long)bn * SEQ * 768 + 384, 768, 24,
                 fub, 384, 24, fub_bf, 384, 24,
                 (const float*)0, SEQ, HD, SEQ, NH, 0, 1.0f, 0, false);
            // out projection + mean over 4 scales into msum[bn/4]
            gemm(fub_bf, 384, 0, wbuf, 384, 0,
                 msum + (long)(bn >> 2) * SEQ * 384, 384, 0, (__bf16*)0, 0, 0,
                 F(bi + 15), SEQ, 384, 384, 1, 0, 0.25f, (bn & 3) != 0, true);
        }
        k_add_ln<<<RWS, 128, 0, stream>>>(h, msum, F(bi + 22), F(bi + 23), h_bf);

        // feed-forward
        cvt(F(bi + 16), wbuf, 1536L * 384);
        gemm(h_bf, 384, 0, wbuf, 384, 0, tmpA, 1536, 0, tmpA_bf, 1536, 0,
             F(bi + 17), 2048, 1536, 384, 1, 1 /*gelu*/, 1.0f, 0, true);
        cvt(F(bi + 18), wbuf, 384L * 1536);
        gemm(tmpA_bf, 1536, 0, wbuf, 1536, 0, tmpB, 384, 0, (__bf16*)0, 0, 0,
             F(bi + 19), 2048, 384, 1536, 1, 0, 1.0f, 0, true);
        k_add_ln<<<RWS, 128, 0, stream>>>(h, tmpB, F(bi + 24), F(bi + 25), h_bf);
    }

    // ---------------- heads ----------------
    k_pool<<<NB * D_MODEL, 128, 0, stream>>>(h, pooled);
    k_heads<<<1, 256, 0, stream>>>(pooled,
        F(135), F(136), F(137), F(138), F(139), F(140),
        F(141), F(142), F(143), F(144), F(145), F(146), F(147), F(148),
        (float*)d_out);
}
